// GraphTransformerLayer_87686052316025
// MI455X (gfx1250) — matrix operand order repaired and checked by reading, not compiled, not run on hardware
//
#include <hip/hip_runtime.h>
#include <hip/hip_bf16.h>
#include <math.h>
#include <stdint.h>

// Problem constants (match reference)
#define Bq    4
#define Nn    512
#define HID   128
#define Hh    8
#define Dd    16
#define Kt    32
#define HQ    32
#define FF    512
#define ROWS  (Bq * Nn)   // 2048

typedef __attribute__((ext_vector_type(16))) __bf16        v16bf;
typedef __attribute__((ext_vector_type(8)))  float         v8f;
typedef __attribute__((ext_vector_type(4)))  unsigned int  v4u;
typedef __attribute__((ext_vector_type(8)))  int           v8i;
typedef __attribute__((ext_vector_type(4)))  int           v4i;

// ---------------------------------------------------------------------------
// Tensor Data Mover helpers (descriptor layout per CDNA5 ISA ch.8)
// ---------------------------------------------------------------------------
__device__ __forceinline__ void tdm_issue(v4u g0, v8i g1, v4i g2, v4i g3)
{
#if defined(__clang_major__) && (__clang_major__ >= 23)
    v8i z8 = {0, 0, 0, 0, 0, 0, 0, 0};
    __builtin_amdgcn_tensor_load_to_lds(g0, g1, g2, g3, z8, 0);
#else
    __builtin_amdgcn_tensor_load_to_lds(g0, g1, g2, g3, 0);
#endif
}

// group0: [1:0]=count=1, [30]=gather_index_size(0 => 16-bit), [31]=gather_mode,
//         [63:32]=lds_addr, [120:64]=global_addr, [127:126]=type=2
__device__ __forceinline__ v4u tdm_g0(unsigned lds_off, unsigned long long ga, bool gather)
{
    v4u g0;
    g0.x = 1u | (gather ? (1u << 31) : 0u);
    g0.y = lds_off;
    g0.z = (unsigned)ga;
    g0.w = (unsigned)((ga >> 32) & 0x1FFFFFFu) | (2u << 30);
    return g0;
}

// group1: data_size=4B (code 2), no barrier/iterate/pad, mask=0,
//         tensor_dim0/1, tile_dim0/1, tensor_dim0_stride (element units)
__device__ __forceinline__ v8i tdm_g1(unsigned td0, unsigned td1,
                                      unsigned tile0, unsigned tile1,
                                      unsigned long long stride0)
{
    v8i g1;
    g1[0] = (int)(2u << 16);                                   // data_size=4B
    g1[1] = (int)((td0 & 0xFFFFu) << 16);                      // ab_addr=0 | td0[15:0]
    g1[2] = (int)((td0 >> 16) | ((td1 & 0xFFFFu) << 16));      // td0[31:16] | td1[15:0]
    g1[3] = (int)((td1 >> 16) | (tile0 << 16));                // td1[31:16] | tile_dim0
    g1[4] = (int)(tile1 & 0xFFFFu);                            // tile_dim1 | tile_dim2=0
    g1[5] = (int)(unsigned)(stride0 & 0xFFFFFFFFull);          // stride0[31:0]
    g1[6] = (int)(unsigned)((stride0 >> 32) & 0xFFFFull);      // stride0[47:32]
    g1[7] = 0;
    return g1;
}

// ---------------------------------------------------------------------------
// Generic GEMM: out[M,Ncol] = act(A[M,Kdim] @ W[Kdim,Ncol] + bias)
// Block = 256 threads = 8 waves -> 32x64 macro-tile (2x4 grid of 16x16 WMMA
// tiles). A/W tiles staged into double-buffered LDS by the TDM (wave 0),
// overlapped with v_wmma_f32_16x16x32_bf16 compute. act: 0=none, 1=gelu.
// Requires M%32==0, Ncol%64==0, Kdim%32==0.
// ---------------------------------------------------------------------------
__global__ __launch_bounds__(256) void gemm_bias_act(
    const float* __restrict__ A, const float* __restrict__ W,
    const float* __restrict__ bias, float* __restrict__ out,
    int M, int Kdim, int Ncol, int act)
{
    __shared__ float sA[2][32][32];   // [buf][m][k]
    __shared__ float sW[2][32][64];   // [buf][k][n]

    const int wave = threadIdx.x >> 5;
    const int lane = threadIdx.x & 31;
    const int wm   = wave >> 2;          // 0..1
    const int wn   = wave & 3;           // 0..3
    const int half = lane >> 4;
    const int ln   = lane & 15;

    const int m0 = blockIdx.y * 32;      // macro-tile row base
    const int n0 = blockIdx.x * 64;      // macro-tile col base
    const int nch = Kdim >> 5;

    const unsigned ldsA = (unsigned)(uintptr_t)&sA[0][0][0];
    const unsigned ldsW = (unsigned)(uintptr_t)&sW[0][0][0];
    const v8i g1A = tdm_g1((unsigned)Kdim, (unsigned)M, 32u, 32u, (unsigned long long)Kdim);
    const v8i g1W = tdm_g1((unsigned)Ncol, (unsigned)Kdim, 64u, 32u, (unsigned long long)Ncol);
    const v4i z4 = {0, 0, 0, 0};

    if (wave == 0) {
        // chunk 0 into buffer 0
        tdm_issue(tdm_g0(ldsA, (unsigned long long)(uintptr_t)(A + (size_t)m0 * Kdim), false), g1A, z4, z4);
        tdm_issue(tdm_g0(ldsW, (unsigned long long)(uintptr_t)(W + n0), false), g1W, z4, z4);
    }

    v8f acc = {};
    for (int kc = 0; kc < nch; ++kc) {
        const int cur = kc & 1;
        if (wave == 0) {
            if (kc + 1 < nch) {
                const int nb = (kc + 1) & 1;
                const int kk = (kc + 1) * 32;
                tdm_issue(tdm_g0(ldsA + (unsigned)nb * 32 * 32 * 4,
                                 (unsigned long long)(uintptr_t)(A + (size_t)m0 * Kdim + kk), false),
                          g1A, z4, z4);
                tdm_issue(tdm_g0(ldsW + (unsigned)nb * 32 * 64 * 4,
                                 (unsigned long long)(uintptr_t)(W + (size_t)kk * Ncol + n0), false),
                          g1W, z4, z4);
                __builtin_amdgcn_s_wait_tensorcnt(2);   // chunk kc arrived
            } else {
                __builtin_amdgcn_s_wait_tensorcnt(0);
            }
        }
        __syncthreads();

        v16bf af, bw;
        #pragma unroll
        for (int j = 0; j < 8; ++j) {
            const int kb = half * 16 + 2 * j;
            af[2 * j]     = (__bf16)sA[cur][wm * 16 + ln][kb];
            af[2 * j + 1] = (__bf16)sA[cur][wm * 16 + ln][kb + 1];
            const int rk = half * 16 + 2 * j;
            bw[2 * j]     = (__bf16)sW[cur][rk][wn * 16 + ln];
            bw[2 * j + 1] = (__bf16)sW[cur][rk + 1][wn * 16 + ln];
        }
        acc = __builtin_amdgcn_wmma_f32_16x16x32_bf16(
                  false, af, false, bw, (short)0, acc, false, false);
        __syncthreads();
    }

    const int col = n0 + wn * 16 + ln;
    const float bv = bias ? bias[col] : 0.0f;
    #pragma unroll
    for (int r = 0; r < 8; ++r) {
        const int m = m0 + wm * 16 + half * 8 + r;
        float v = acc[r] + bv;
        if (act == 1) v = 0.5f * v * (1.0f + erff(v * 0.70710678118654752f));
        out[(size_t)m * Ncol + col] = v;
    }
}

// ---------------------------------------------------------------------------
// Top-k (k=32) smallest-metric neighbor selection. One wave per (b,q) row.
// metric = (allowed ? spd : 1e6) + key * 1e-4
// ---------------------------------------------------------------------------
__global__ __launch_bounds__(32) void topk_kernel(
    const float* __restrict__ spd,
    const unsigned char* __restrict__ node_mask,
    const unsigned char* __restrict__ edge_mask,
    int* __restrict__ nn_idx)
{
    const int row  = blockIdx.x;           // b*N + q
    const int b    = row >> 9;
    const int q    = row & (Nn - 1);
    const int lane = threadIdx.x;

    const float* __restrict__ sp = spd + (size_t)row * Nn;
    const unsigned char* __restrict__ em = edge_mask + (size_t)row * Nn;
    const unsigned char* __restrict__ nm = node_mask + (size_t)b * Nn;

    float metric[16];
    #pragma unroll
    for (int t = 0; t < 16; ++t) {
        const int key = t * 32 + lane;
        bool allowed = (em[key] != 0) || (key == q) || (key == 0);
        allowed = allowed && (nm[key] != 0);
        metric[t] = (allowed ? sp[key] : 1.0e6f) + (float)key * 1.0e-4f;
    }

    unsigned taken = 0;
    for (int sel = 0; sel < Kt; ++sel) {
        float bv = 3.0e38f;
        int   bi = 1 << 30;
        #pragma unroll
        for (int t = 0; t < 16; ++t) {
            if (!((taken >> t) & 1u) && metric[t] < bv) {
                bv = metric[t];
                bi = t * 32 + lane;
            }
        }
        #pragma unroll
        for (int off = 16; off > 0; off >>= 1) {
            const float ov = __shfl_xor(bv, off, 32);
            const int   oi = __shfl_xor(bi, off, 32);
            if (ov < bv || (ov == bv && oi < bi)) { bv = ov; bi = oi; }
        }
        if (lane == 0) nn_idx[(size_t)row * Kt + sel] = bi;
        if ((bi & 31) == lane) taken |= 1u << (bi >> 5);
    }
}

// ---------------------------------------------------------------------------
// Fused attention. One 128-thread block (4 waves) per (b,q) row.
// e_sel gathered by TDM gather-mode (2 descriptors x 16 row indices),
// overlapped with the raw score computation; edge-bias MLP via WMMA.
// ---------------------------------------------------------------------------
__global__ __launch_bounds__(128) void attn_kernel(
    const float* __restrict__ Qb, const float* __restrict__ Kb,
    const float* __restrict__ Vb,
    const float* __restrict__ e,  const float* __restrict__ spd,
    const unsigned char* __restrict__ node_mask,
    const unsigned char* __restrict__ edge_mask,
    const int* __restrict__ nn_idx,
    const float* __restrict__ eb_w1, const float* __restrict__ eb_b1,
    const float* __restrict__ eb_g,  const float* __restrict__ eb_beta,
    const float* __restrict__ eb_w2, const float* __restrict__ eb_b2,
    const float* __restrict__ sb_w1, const float* __restrict__ sb_b1,
    const float* __restrict__ sb_w2, const float* __restrict__ sb_b2,
    float* __restrict__ attn_out)
{
    __shared__ int   s_idx[Kt];
    __shared__ float s_esel[Kt][HID];      // 16 KB gathered edge features (TDM)
    __shared__ float s_x[Kt][HQ + 1];      // pre-LN edge MLP hidden
    __shared__ float s_scores[Hh][Kt];
    __shared__ float s_be[Kt][Hh];
    __shared__ float s_bs[Kt][Hh];
    __shared__ float s_attn[Hh][Kt];

    const int row = blockIdx.x;            // b*N + q
    const int b   = row >> 9;
    const int tid = threadIdx.x;

    if (tid < Kt) s_idx[tid] = nn_idx[(size_t)row * Kt + tid];
    __syncthreads();

    // wave 0: issue two TDM gather descriptors pulling the 32 selected rows of
    // e[b,q,:,:] straight into LDS (16 x 16-bit indices per descriptor)
    if (tid < 32) {
        const unsigned lds0 = (unsigned)(uintptr_t)&s_esel[0][0];
        const unsigned long long ga =
            (unsigned long long)(uintptr_t)(e + (size_t)row * Nn * HID);
        const v8i g1 = tdm_g1(HID, Nn, HID, 16u, HID);
        #pragma unroll
        for (int hgrp = 0; hgrp < 2; ++hgrp) {
            v4i g2, g3;
            #pragma unroll
            for (int j = 0; j < 4; ++j) {
                g2[j] = (s_idx[hgrp * 16 + 2 * j] & 0xFFFF) |
                        (s_idx[hgrp * 16 + 2 * j + 1] << 16);
                g3[j] = (s_idx[hgrp * 16 + 8 + 2 * j] & 0xFFFF) |
                        (s_idx[hgrp * 16 + 8 + 2 * j + 1] << 16);
            }
            tdm_issue(tdm_g0(lds0 + (unsigned)hgrp * 16 * HID * 4, ga, true), g1, g2, g3);
        }
    }

    // overlap with TDM: raw attention scores (Q . K_sel) / sqrt(D)
    for (int i = tid; i < Hh * Kt; i += 128) {
        const int h = i >> 5, k = i & 31;
        const int key = s_idx[k];
        const float* __restrict__ qp = Qb + (size_t)row * HID + h * Dd;
        const float* __restrict__ kp = Kb + ((size_t)(b * Nn) + key) * HID + h * Dd;
        float s = 0.0f;
        #pragma unroll
        for (int d = 0; d < Dd; ++d) s += qp[d] * kp[d];
        s_scores[h][k] = s * 0.25f;        // 1/sqrt(16)
    }

    if (tid < 32) __builtin_amdgcn_s_wait_tensorcnt(0);
    __syncthreads();

    // edge-bias hidden: x = e_sel(32x128) @ eb_w1(128x32) + b1  (WMMA, 2x2 tiles)
    {
        const int wave = tid >> 5, lane = tid & 31;
        const int mt = wave >> 1, nt = wave & 1;
        const int half = lane >> 4, ln = lane & 15;
        v8f acc = {};
        for (int kk = 0; kk < HID; kk += 32) {
            v16bf af, bw;
            #pragma unroll
            for (int j = 0; j < 8; ++j) {
                const int kb = kk + half * 16 + 2 * j;
                af[2 * j]     = (__bf16)s_esel[mt * 16 + ln][kb];
                af[2 * j + 1] = (__bf16)s_esel[mt * 16 + ln][kb + 1];
                const int rk = kk + half * 16 + 2 * j;
                bw[2 * j]     = (__bf16)eb_w1[rk * HQ + nt * 16 + ln];
                bw[2 * j + 1] = (__bf16)eb_w1[(rk + 1) * HQ + nt * 16 + ln];
            }
            acc = __builtin_amdgcn_wmma_f32_16x16x32_bf16(
                      false, af, false, bw, (short)0, acc, false, false);
        }
        #pragma unroll
        for (int r = 0; r < 8; ++r) {
            const int m = mt * 16 + half * 8 + r;
            const int c = nt * 16 + ln;
            s_x[m][c] = acc[r] + eb_b1[c];
        }
    }
    __syncthreads();

    // per-neighbor: LN + leaky + eb_w2  and  spd-bias MLP
    if (tid < Kt) {
        const int k = tid;
        float mu = 0.0f;
        #pragma unroll
        for (int c = 0; c < HQ; ++c) mu += s_x[k][c];
        mu *= (1.0f / HQ);
        float var = 0.0f;
        #pragma unroll
        for (int c = 0; c < HQ; ++c) { const float d = s_x[k][c] - mu; var += d * d; }
        const float inv = rsqrtf(var * (1.0f / HQ) + 1.0e-5f);
        float xn[HQ];
        #pragma unroll
        for (int c = 0; c < HQ; ++c) {
            float v = (s_x[k][c] - mu) * inv * eb_g[c] + eb_beta[c];
            xn[c] = (v >= 0.0f) ? v : 0.2f * v;
        }
        #pragma unroll
        for (int hh = 0; hh < Hh; ++hh) {
            float s = eb_b2[hh];
            #pragma unroll
            for (int c = 0; c < HQ; ++c) s += xn[c] * eb_w2[c * Hh + hh];
            s_be[k][hh] = s;
        }
        // spd bias: scalar -> HQ -> H
        const int key = s_idx[k];
        const float sv = spd[(size_t)row * Nn + key];
        const float sn = fminf(fmaxf(sv, 0.0f), 10.0f) * 0.1f;
        float y[HQ];
        #pragma unroll
        for (int c = 0; c < HQ; ++c) {
            float v = sn * sb_w1[c] + sb_b1[c];
            y[c] = (v >= 0.0f) ? v : 0.2f * v;
        }
        #pragma unroll
        for (int hh = 0; hh < Hh; ++hh) {
            float s = sb_b2[hh];
            #pragma unroll
            for (int c = 0; c < HQ; ++c) s += y[c] * sb_w2[c * Hh + hh];
            s_bs[k][hh] = s;
        }
    }
    __syncthreads();

    // combine biases, mask invalid keys, softmax per head
    if (tid < Hh) {
        const int h = tid;
        float sc[Kt];
        float mx = -3.0e38f;
        #pragma unroll
        for (int k = 0; k < Kt; ++k) {
            const int key = s_idx[k];
            const bool es = edge_mask[(size_t)row * Nn + key] != 0;
            float s = s_scores[h][k] + (es ? s_be[k][h] : s_bs[k][h]);
            const bool kv = node_mask[(size_t)b * Nn + key] != 0;
            s = kv ? s : -1.0e9f;
            sc[k] = s;
            mx = fmaxf(mx, s);
        }
        float sum = 0.0f;
        #pragma unroll
        for (int k = 0; k < Kt; ++k) { sc[k] = expf(sc[k] - mx); sum += sc[k]; }
        const float r = 1.0f / sum;
        #pragma unroll
        for (int k = 0; k < Kt; ++k) s_attn[h][k] = sc[k] * r;
    }
    __syncthreads();

    // out[h][d] = sum_k attn * V_sel  (128 outputs, 1 per thread)
    {
        const int h = tid >> 4, d = tid & 15;
        float s = 0.0f;
        #pragma unroll
        for (int k = 0; k < Kt; ++k) {
            const int key = s_idx[k];
            s += s_attn[h][k] * Vb[((size_t)(b * Nn) + key) * HID + h * Dd + d];
        }
        attn_out[(size_t)row * HID + h * Dd + d] = s;
    }
}

// ---------------------------------------------------------------------------
// out = LayerNorm(X + R) * g + beta, 128-wide rows, one wave per row.
// ---------------------------------------------------------------------------
__global__ __launch_bounds__(256) void add_ln_kernel(
    const float* __restrict__ X, const float* __restrict__ R,
    const float* __restrict__ g, const float* __restrict__ beta,
    float* __restrict__ out, int Mrows)
{
    const int wave = threadIdx.x >> 5;
    const int lane = threadIdx.x & 31;
    const int row  = blockIdx.x * 8 + wave;
    if (row >= Mrows) return;

    const float* __restrict__ xp = X + (size_t)row * HID;
    const float* __restrict__ rp = R + (size_t)row * HID;
    float v[4];
    float sum = 0.0f;
    #pragma unroll
    for (int j = 0; j < 4; ++j) {
        v[j] = xp[lane + 32 * j] + rp[lane + 32 * j];
        sum += v[j];
    }
    #pragma unroll
    for (int off = 16; off > 0; off >>= 1) sum += __shfl_xor(sum, off, 32);
    const float mu = sum * (1.0f / HID);
    float vs = 0.0f;
    #pragma unroll
    for (int j = 0; j < 4; ++j) { const float d = v[j] - mu; vs += d * d; }
    #pragma unroll
    for (int off = 16; off > 0; off >>= 1) vs += __shfl_xor(vs, off, 32);
    const float inv = rsqrtf(vs * (1.0f / HID) + 1.0e-5f);
    #pragma unroll
    for (int j = 0; j < 4; ++j) {
        const int c = lane + 32 * j;
        out[(size_t)row * HID + c] = (v[j] - mu) * inv * g[c] + beta[c];
    }
}

// ---------------------------------------------------------------------------
extern "C" void kernel_launch(void* const* d_in, const int* in_sizes, int n_in,
                              void* d_out, int out_size, void* d_ws, size_t ws_size,
                              hipStream_t stream)
{
    const float* h         = (const float*)d_in[0];
    const float* e         = (const float*)d_in[1];
    const float* spd       = (const float*)d_in[2];
    const unsigned char* node_mask = (const unsigned char*)d_in[3];
    const unsigned char* edge_mask = (const unsigned char*)d_in[4];
    const float* wq   = (const float*)d_in[5];
    const float* wk   = (const float*)d_in[6];
    const float* wv   = (const float*)d_in[7];
    const float* wo   = (const float*)d_in[8];
    const float* wo_b = (const float*)d_in[9];
    const float* eb_w1   = (const float*)d_in[10];
    const float* eb_b1   = (const float*)d_in[11];
    const float* eb_g    = (const float*)d_in[12];
    const float* eb_beta = (const float*)d_in[13];
    const float* eb_w2   = (const float*)d_in[14];
    const float* eb_b2   = (const float*)d_in[15];
    const float* sb_w1   = (const float*)d_in[16];
    const float* sb_b1   = (const float*)d_in[17];
    const float* sb_w2   = (const float*)d_in[18];
    const float* sb_b2   = (const float*)d_in[19];
    const float* n1_g = (const float*)d_in[20];
    const float* n1_b = (const float*)d_in[21];
    const float* n2_g = (const float*)d_in[22];
    const float* n2_b = (const float*)d_in[23];
    const float* f1_w = (const float*)d_in[24];
    const float* f1_b = (const float*)d_in[25];
    const float* f2_w = (const float*)d_in[26];
    const float* f2_b = (const float*)d_in[27];

    // workspace layout (floats)
    float* ws = (float*)d_ws;
    const size_t RH = (size_t)ROWS * HID;       // 262144
    float* Qb       = ws;
    float* Kb       = Qb + RH;
    float* Vb       = Kb + RH;
    float* attn_out = Vb + RH;
    float* proj     = attn_out + RH;
    float* h1       = proj + RH;
    float* ff1      = h1 + RH;                  // ROWS*FF = 1048576
    float* ff2      = ff1 + (size_t)ROWS * FF;
    int*   nn       = (int*)(ff2 + RH);

    const dim3 blk256(256);
    auto ggrid = [](int M, int Ncol) { return dim3(Ncol / 64, M / 32); };

    // Q, K, V projections
    gemm_bias_act<<<ggrid(ROWS, HID), blk256, 0, stream>>>(h, wq, nullptr, Qb, ROWS, HID, HID, 0);
    gemm_bias_act<<<ggrid(ROWS, HID), blk256, 0, stream>>>(h, wk, nullptr, Kb, ROWS, HID, HID, 0);
    gemm_bias_act<<<ggrid(ROWS, HID), blk256, 0, stream>>>(h, wv, nullptr, Vb, ROWS, HID, HID, 0);

    // neighbor selection
    topk_kernel<<<ROWS, dim3(32), 0, stream>>>(spd, node_mask, edge_mask, nn);

    // fused sparse attention
    attn_kernel<<<ROWS, dim3(128), 0, stream>>>(Qb, Kb, Vb, e, spd, node_mask, edge_mask, nn,
                                                eb_w1, eb_b1, eb_g, eb_beta, eb_w2, eb_b2,
                                                sb_w1, sb_b1, sb_w2, sb_b2, attn_out);

    // output projection + residual LN
    gemm_bias_act<<<ggrid(ROWS, HID), blk256, 0, stream>>>(attn_out, wo, wo_b, proj, ROWS, HID, HID, 0);
    add_ln_kernel<<<ROWS / 8, blk256, 0, stream>>>(h, proj, n1_g, n1_b, h1, ROWS);

    // FFN: gelu(h1 @ f1 + b1) @ f2 + b2, then residual LN into d_out
    gemm_bias_act<<<ggrid(ROWS, FF), blk256, 0, stream>>>(h1, f1_w, f1_b, ff1, ROWS, HID, FF, 1);
    gemm_bias_act<<<ggrid(ROWS, HID), blk256, 0, stream>>>(ff1, f2_w, f2_b, ff2, ROWS, FF, HID, 0);
    add_ln_kernel<<<ROWS / 8, blk256, 0, stream>>>(h1, ff2, n2_g, n2_b, (float*)d_out, ROWS);
}
